// Top_HiCL_Matching_9612136808769
// MI455X (gfx1250) — matrix operand order repaired and checked by reading, not compiled, not run on hardware
//
#include <hip/hip_runtime.h>
#include <hip/hip_bf16.h>

// ---------------- problem constants (from reference) ----------------
#define N_JOBS   100000
#define N_SKILLS 30000
#define DIM      256
#define B_USERS  2048
#define L_SK     50
#define INV_TEMP 10.0f                       // 1/TEMP
#define SCALE2   14.4269504088896340f        // INV_TEMP * log2(e)  (base-2 logits)
#define LN2      0.6931471805599453f

#define J_SLABS  25                          // 100000/25 = 4000 jobs/slab = 250 N-tiles
#define ROW_TILES (B_USERS / 16)             // 128
#define TILES_PER_SLAB ((N_JOBS / 16) / J_SLABS)  // 250 (even -> clean pairs)
#define WAVES    8

typedef __attribute__((ext_vector_type(16))) __bf16 bf16x16;
typedef __attribute__((ext_vector_type(8)))  __bf16 bf16x8;
typedef __attribute__((ext_vector_type(8)))  float  v8f;

// ---------------- kernel 0: E_j f32 -> bf16 ----------------
__global__ void cvt_ej_bf16(const float* __restrict__ Ej, __bf16* __restrict__ Ejb) {
    long i = ((long)blockIdx.x * blockDim.x + threadIdx.x) * 4;   // N_JOBS*DIM % 1024 == 0
    float4 v = *(const float4*)(Ej + i);
    Ejb[i + 0] = (__bf16)v.x;
    Ejb[i + 1] = (__bf16)v.y;
    Ejb[i + 2] = (__bf16)v.z;
    Ejb[i + 3] = (__bf16)v.w;
}

// ---------------- kernel 1: ragged mean-pool -> bf16, plus f32 target logit ----------------
__global__ void pool_kernel(const float* __restrict__ Es,
                            const int* __restrict__ skill_ids,
                            const int* __restrict__ lengths,
                            const float* __restrict__ Ej,
                            const int* __restrict__ user_jobs,
                            __bf16* __restrict__ Pb,
                            float* __restrict__ tgt) {
    const int b = blockIdx.x;        // one user per block
    const int d = threadIdx.x;       // 256 threads == DIM
    const int len = lengths[b];
    const int* ids = skill_ids + b * L_SK;
    float acc = 0.0f;
    for (int l = 0; l < len; ++l)
        acc += Es[(long)ids[l] * DIM + d];
    const float pv = acc / (float)len;
    Pb[b * DIM + d] = (__bf16)pv;

    // target logit in f32: dot(pooled, E_j[user_jobs[b]]) * INV_TEMP
    __shared__ float red[DIM];
    red[d] = pv * Ej[(long)user_jobs[b] * DIM + d];
    __syncthreads();
    for (int s = DIM / 2; s > 0; s >>= 1) {
        if (d < s) red[d] += red[d + s];
        __syncthreads();
    }
    if (d == 0) tgt[b] = red[0] * INV_TEMP;
}

// ---------------- WMMA helper: full K=256 16x16 logit tile ----------------
__device__ __forceinline__ v8f wmma_tile(const bf16x16 (&afrag)[8],
                                         const __bf16* __restrict__ brow) {
    v8f acc = {};
    #pragma unroll
    for (int c = 0; c < 8; ++c) {
        const __bf16* p = brow + 32 * c;
        bf16x8 lo = *(const bf16x8*)(p);
        bf16x8 hi = *(const bf16x8*)(p + 16);
        bf16x16 bfrag;
        #pragma unroll
        for (int j = 0; j < 8; ++j) { bfrag[j] = lo[j]; bfrag[8 + j] = hi[j]; }
        acc = __builtin_amdgcn_wmma_f32_16x16x32_bf16(
            false, afrag[c], false, bfrag, (short)0, acc, false, false);
    }
    return acc;
}

// ---------------- kernel 2: fused bf16-WMMA GEMM + per-lane online LSE ----------------
// grid: (J_SLABS, ROW_TILES), block 256 (8 waves). Hot loop has NO cross-lane ops:
// each lane keeps private running (max, sumexp) per row; lanes merge once at the end.
__global__ void __launch_bounds__(256)
gemm_lse_kernel(const __bf16* __restrict__ Ejb,
                const __bf16* __restrict__ Pb,
                float* __restrict__ pmax,
                float* __restrict__ psum) {
    const int slab = blockIdx.x;
    const int rt   = blockIdx.y;
    const int lane = threadIdx.x & 31;
    const int wave = threadIdx.x >> 5;
    const int half = lane >> 4;      // 0 or 1
    const int lr   = lane & 15;      // 0..15
    const int rowbase = rt * 16;

    // --- A fragments: pooled rows, 8 K-chunks of 32 ---
    // 16-bit A 16x32 layout: lane holds K = 32c + half*8 + j (j=0..7) and +16.
    bf16x16 afrag[8];
    {
        const __bf16* arow = Pb + (rowbase + lr) * DIM;
        #pragma unroll
        for (int c = 0; c < 8; ++c) {
            const __bf16* p = arow + 32 * c + half * 8;
            bf16x8 lo = *(const bf16x8*)(p);
            bf16x8 hi = *(const bf16x8*)(p + 16);
            #pragma unroll
            for (int j = 0; j < 8; ++j) { afrag[c][j] = lo[j]; afrag[c][8 + j] = hi[j]; }
        }
    }

    float rmax[8], rsum[8];
    #pragma unroll
    for (int v = 0; v < 8; ++v) { rmax[v] = -3.0e38f; rsum[v] = 0.0f; }

    const int tile0 = slab * TILES_PER_SLAB;
    // two N-tiles per iteration: 16 WMMAs per epilogue, rescale amortized
    for (int tt = wave * 2; tt < TILES_PER_SLAB; tt += WAVES * 2) {
        const int jb0 = (tile0 + tt) * 16;
        // 16-bit B 32x16 layout: lane = N column, K = 32c + half*8 + j (j=0..7) and +16.
        const __bf16* brow0 = Ejb + (long)(jb0 + lr) * DIM + half * 8;
        const __bf16* brow1 = brow0 + 16 * DIM;
        v8f acc0 = wmma_tile(afrag, brow0);
        v8f acc1 = wmma_tile(afrag, brow1);
        #pragma unroll
        for (int v = 0; v < 8; ++v) {
            const float x0 = acc0[v] * SCALE2;     // base-2 logits
            const float x1 = acc1[v] * SCALE2;
            const float mt = fmaxf(x0, x1);
            const float nm = fmaxf(rmax[v], mt);
            rsum[v] = rsum[v] * exp2f(rmax[v] - nm) + exp2f(x0 - nm) + exp2f(x1 - nm);
            rmax[v] = nm;
        }
    }

    // --- cross-lane merge (once per wave): 16 lanes of each row group -> one (m, s) ---
    #pragma unroll
    for (int v = 0; v < 8; ++v) {
        float m = rmax[v], s = rsum[v];
        #pragma unroll
        for (int o = 8; o >= 1; o >>= 1) {
            const float om = __shfl_xor(m, o, 16);
            const float os = __shfl_xor(s, o, 16);
            const float nm = fmaxf(m, om);
            s = s * exp2f(m - nm) + os * exp2f(om - nm);
            m = nm;
        }
        rmax[v] = m; rsum[v] = s;
    }

    // --- merge the 8 waves via LDS ---
    __shared__ float smax[WAVES][16], ssum[WAVES][16];
    if (lr == 0) {
        #pragma unroll
        for (int v = 0; v < 8; ++v) {
            const int r = half * 8 + v;
            smax[wave][r] = rmax[v];
            ssum[wave][r] = rsum[v];
        }
    }
    __syncthreads();
    if (threadIdx.x < 16) {
        const int r = threadIdx.x;
        float M = -3.0e38f;
        #pragma unroll
        for (int w = 0; w < WAVES; ++w) M = fmaxf(M, smax[w][r]);
        float S = 0.0f;
        #pragma unroll
        for (int w = 0; w < WAVES; ++w) S += ssum[w][r] * exp2f(smax[w][r] - M);
        const int b = rowbase + r;
        pmax[b * J_SLABS + slab] = M;     // base-2 scale
        psum[b * J_SLABS + slab] = S;
    }
}

// ---------------- kernel 3: merge slabs per user, NLL mean ----------------
__global__ void finalize_kernel(const float* __restrict__ pmax,
                                const float* __restrict__ psum,
                                const float* __restrict__ tgt,
                                float* __restrict__ out) {
    __shared__ float red[256];
    float local = 0.0f;
    for (int b = threadIdx.x; b < B_USERS; b += 256) {
        float M = -3.0e38f;
        for (int j = 0; j < J_SLABS; ++j) M = fmaxf(M, pmax[b * J_SLABS + j]);
        float S = 0.0f;
        for (int j = 0; j < J_SLABS; ++j) S += psum[b * J_SLABS + j] * exp2f(pmax[b * J_SLABS + j] - M);
        local += LN2 * (M + log2f(S)) - tgt[b];   // logsumexp(nat) - target_logit(nat)
    }
    red[threadIdx.x] = local;
    __syncthreads();
    for (int s = 128; s > 0; s >>= 1) {
        if (threadIdx.x < s) red[threadIdx.x] += red[threadIdx.x + s];
        __syncthreads();
    }
    if (threadIdx.x == 0) out[0] = red[0] / (float)B_USERS;
}

// ---------------- launcher ----------------
extern "C" void kernel_launch(void* const* d_in, const int* in_sizes, int n_in,
                              void* d_out, int out_size, void* d_ws, size_t ws_size,
                              hipStream_t stream) {
    const float* Ej        = (const float*)d_in[0];   // [N_JOBS, DIM]
    const float* Es        = (const float*)d_in[1];   // [N_SKILLS, DIM]
    const int*   user_jobs = (const int*)d_in[2];     // [B]
    const int*   skill_ids = (const int*)d_in[3];     // [B, L]
    const int*   lengths   = (const int*)d_in[4];     // [B]
    float* out = (float*)d_out;

    // workspace carve-up (256B aligned)
    char* ws = (char*)d_ws;
    size_t off = 0;
    __bf16* Ejb = (__bf16*)(ws + off); off += (size_t)N_JOBS * DIM * sizeof(__bf16);
    off = (off + 255) & ~(size_t)255;
    __bf16* Pb  = (__bf16*)(ws + off); off += (size_t)B_USERS * DIM * sizeof(__bf16);
    off = (off + 255) & ~(size_t)255;
    float* pmax = (float*)(ws + off);  off += (size_t)B_USERS * J_SLABS * sizeof(float);
    float* psum = (float*)(ws + off);  off += (size_t)B_USERS * J_SLABS * sizeof(float);
    float* tgtv = (float*)(ws + off);  off += (size_t)B_USERS * sizeof(float);

    // 0) E_j -> bf16
    cvt_ej_bf16<<<(N_JOBS * DIM) / (256 * 4), 256, 0, stream>>>(Ej, Ejb);
    // 1) ragged mean-pool -> bf16 + f32 target logits
    pool_kernel<<<B_USERS, DIM, 0, stream>>>(Es, skill_ids, lengths, Ej, user_jobs, Pb, tgtv);
    // 2) fused WMMA GEMM + per-lane online logsumexp
    gemm_lse_kernel<<<dim3(J_SLABS, ROW_TILES), 256, 0, stream>>>(Ejb, Pb, pmax, psum);
    // 3) merge + mean NLL
    finalize_kernel<<<1, 256, 0, stream>>>(pmax, psum, tgtv, out);
}
